// SelfAttention1d_12429635354828
// MI455X (gfx1250) — matrix operand order repaired and checked by reading, not compiled, not run on hardware
//
#include <hip/hip_runtime.h>
#include <stdint.h>

// Problem constants (from reference)
#define BB 8
#define CC 512
#define NN 2048
#define DD 64   // CQK

typedef __attribute__((ext_vector_type(16))) __bf16 bf16x16;
typedef __attribute__((ext_vector_type(8)))  float  f32x8;

union Frag32B { uint4 u[2]; bf16x16 v; };

__device__ __forceinline__ uint16_t f2bf(float f) {
    uint32_t u = __float_as_uint(f);
    uint32_t r = u + 0x7FFFu + ((u >> 16) & 1u);   // round-to-nearest-even
    return (uint16_t)(r >> 16);
}

// A-fragment (16x32 bf16, MxK): lane m = l&15, lane-half picks k sub-block.
// p points at row 0 of the 16-row tile, ld in elements, rows contiguous in k.
__device__ __forceinline__ bf16x16 load_a(const uint16_t* p, int ld, int row, int kb, int hi) {
    const uint16_t* q = p + (size_t)row * ld + kb + hi * 8;
    Frag32B f;
    f.u[0] = *(const uint4*)(q);        // V0-3: k = kb + hi*8 + 0..7
    f.u[1] = *(const uint4*)(q + 16);   // V4-7: k = kb + 16 + hi*8 + 0..7
    return f.v;
}

// B-fragment (32x16 bf16, KxN) from "Bt" storage: row n, k contiguous.
// lane n = l&15, lane-half picks k sub-block (same order as A).
__device__ __forceinline__ bf16x16 load_b(const uint16_t* p, int ld, int nrow, int kb, int hi) {
    const uint16_t* q = p + (size_t)nrow * ld + kb + hi * 8;
    Frag32B f;
    f.u[0] = *(const uint4*)(q);        // V0-3: k = kb + hi*8 + 0..7
    f.u[1] = *(const uint4*)(q + 16);   // V4-7: k = kb + 16 + hi*8 + 0..7
    return f.v;
}

__device__ __forceinline__ f32x8 wmma_bf16(bf16x16 a, bf16x16 b, f32x8 c) {
    return __builtin_amdgcn_wmma_f32_16x16x32_bf16(
        false, a, false, b, (short)0, c, false, false);
}

// ---------------- elementwise fp32 -> bf16 convert (weights) ----------------
__global__ void cvt_f32_bf16(const float* __restrict__ src, uint16_t* __restrict__ dst, int n) {
    int i = blockIdx.x * blockDim.x + threadIdx.x;
    if (i < n) dst[i] = f2bf(src[i]);
}

// -------- tiled transpose+convert: x[B,C,N] f32 -> xT[B,N,C] bf16 ----------
__global__ void transpose_x(const float* __restrict__ x, uint16_t* __restrict__ xT) {
    __shared__ float tile[32][33];
    int b  = blockIdx.z;
    int n0 = blockIdx.x * 32;
    int c0 = blockIdx.y * 32;
    int tx = threadIdx.x, ty = threadIdx.y;   // 32 x 8
    const float* xb = x + (size_t)b * CC * NN;
#pragma unroll
    for (int k = 0; k < 32; k += 8)
        tile[ty + k][tx] = xb[(size_t)(c0 + ty + k) * NN + n0 + tx];
    __syncthreads();
    uint16_t* xTb = xT + (size_t)b * NN * CC;
#pragma unroll
    for (int k = 0; k < 32; k += 8)
        xTb[(size_t)(n0 + ty + k) * CC + c0 + tx] = f2bf(tile[tx][ty + k]);
}

// ---- q/k projection: waves 0-3 -> q, waves 4-7 -> k; both stored [B,N,64] ----
__global__ void proj_qk(const uint16_t* __restrict__ xT,
                        const uint16_t* __restrict__ Wqb, const float* __restrict__ bq,
                        const uint16_t* __restrict__ Wkb, const float* __restrict__ bk,
                        uint16_t* __restrict__ qO, uint16_t* __restrict__ kO) {
    int blk = blockIdx.x;                 // B * (N/64)
    int b   = blk / (NN / 64);
    int n0  = (blk % (NN / 64)) * 64;
    int tid = threadIdx.x;
    int w = tid >> 5, l = tid & 31, lane16 = l & 15, hi = (l >> 4) & 1;
    const uint16_t* W    = (w < 4) ? Wqb : Wkb;
    const float*    bias = (w < 4) ? bq  : bk;
    uint16_t*       out  = (w < 4) ? qO  : kO;
    int d0 = (w & 3) * 16;
    const uint16_t* xTb = xT + (size_t)b * NN * CC;

    f32x8 acc[4];
#pragma unroll
    for (int nt = 0; nt < 4; nt++) acc[nt] = (f32x8){0,0,0,0,0,0,0,0};

    for (int kb = 0; kb < CC; kb += 32) {
        bf16x16 a = load_a(W, CC, d0 + lane16, kb, hi);
#pragma unroll
        for (int nt = 0; nt < 4; nt++) {
            bf16x16 bf = load_b(xTb, CC, n0 + nt * 16 + lane16, kb, hi);
            acc[nt] = wmma_bf16(a, bf, acc[nt]);
        }
    }
#pragma unroll
    for (int nt = 0; nt < 4; nt++) {
        int i = n0 + nt * 16 + lane16;   // position (D-frag col)
#pragma unroll
        for (int r = 0; r < 8; r++) {
            int d = d0 + r + hi * 8;     // channel (D-frag row)
            out[((size_t)b * NN + i) * DD + d] = f2bf(acc[nt][r] + bias[d]);
        }
    }
}

// ---- v projection: v[B,C,N] bf16 (bias folded in) ----
__global__ void proj_v(const uint16_t* __restrict__ xT,
                       const uint16_t* __restrict__ Wvb, const float* __restrict__ bv,
                       uint16_t* __restrict__ vO) {
    int blk = blockIdx.x;                 // B * (C/64) * (N/128)
    int nb  = blk % (NN / 128); blk /= (NN / 128);
    int cb  = blk % (CC / 64);
    int b   = blk / (CC / 64);
    int tid = threadIdx.x;
    int w = tid >> 5, l = tid & 31, lane16 = l & 15, hi = (l >> 4) & 1;
    int c0 = cb * 64 + (w & 3) * 16;
    int n0 = nb * 128 + (w >> 2) * 64;
    const uint16_t* xTb = xT + (size_t)b * NN * CC;

    f32x8 acc[4];
#pragma unroll
    for (int nt = 0; nt < 4; nt++) acc[nt] = (f32x8){0,0,0,0,0,0,0,0};

    for (int kb = 0; kb < CC; kb += 32) {
        bf16x16 a = load_a(Wvb, CC, c0 + lane16, kb, hi);
#pragma unroll
        for (int nt = 0; nt < 4; nt++) {
            bf16x16 bf = load_b(xTb, CC, n0 + nt * 16 + lane16, kb, hi);
            acc[nt] = wmma_bf16(a, bf, acc[nt]);
        }
    }
#pragma unroll
    for (int nt = 0; nt < 4; nt++) {
        int n = n0 + nt * 16 + lane16;
#pragma unroll
        for (int r = 0; r < 8; r++) {
            int c = c0 + r + hi * 8;
            vO[((size_t)b * CC + c) * NN + n] = f2bf(acc[nt][r] + bv[c]);
        }
    }
}

// ---- fused attention: energy WMMA -> block softmax -> P(LDS) -> out WMMA ----
// grid = B * (N/16), 256 threads (8 waves). Wave w: energy j-slice w*256,
// then output channels c = w*64 .. w*64+63.
__global__ void attn(const uint16_t* __restrict__ qA, const uint16_t* __restrict__ kA,
                     const uint16_t* __restrict__ vA, const float* __restrict__ x,
                     const float* __restrict__ gamma, float* __restrict__ out) {
    extern __shared__ char smem[];
    uint16_t* P      = (uint16_t*)smem;                 // [16][2048] bf16 = 64KB
    float*    redmax = (float*)(smem + 16 * NN * 2);    // [8 waves][16 rows]
    float*    redsum = redmax + 128;                    // [8 waves][16 rows]

    int blk = blockIdx.x;
    int b   = blk / (NN / 16);
    int i0  = (blk % (NN / 16)) * 16;
    int tid = threadIdx.x;
    int w = tid >> 5, l = tid & 31, lane16 = l & 15, hi = (l >> 4) & 1;

    // ---------------- pass 1: energy for j in [j0, j0+256) ----------------
    const uint16_t* qb = qA + ((size_t)b * NN + i0) * DD;
    bf16x16 aq0 = load_a(qb, DD, lane16, 0,  hi);   // K = 0..31  (d)
    bf16x16 aq1 = load_a(qb, DD, lane16, 32, hi);   // K = 32..63
    const uint16_t* kb_ = kA + (size_t)b * NN * DD; // k stored [N,64] like q
    int j0 = w * 256;

    f32x8 accE[16];
#pragma unroll
    for (int nt = 0; nt < 16; nt++) {
        int j = j0 + nt * 16;
        bf16x16 b0 = load_b(kb_, DD, j + lane16, 0,  hi);
        bf16x16 b1 = load_b(kb_, DD, j + lane16, 32, hi);
        f32x8 e = (f32x8){0,0,0,0,0,0,0,0};
        e = wmma_bf16(aq0, b0, e);
        e = wmma_bf16(aq1, b1, e);
        accE[nt] = e;
    }

    // local row max (row m = r + hi*8), reduce across this wave's 16 n-tiles
    float pm[8];
#pragma unroll
    for (int r = 0; r < 8; r++) pm[r] = accE[0][r];
#pragma unroll
    for (int nt = 1; nt < 16; nt++)
#pragma unroll
        for (int r = 0; r < 8; r++) pm[r] = fmaxf(pm[r], accE[nt][r]);
#pragma unroll
    for (int m = 1; m < 16; m <<= 1)
#pragma unroll
        for (int r = 0; r < 8; r++) pm[r] = fmaxf(pm[r], __shfl_xor(pm[r], m, 32));
    if (lane16 == 0) {
#pragma unroll
        for (int r = 0; r < 8; r++) redmax[w * 16 + r + hi * 8] = pm[r];
    }
    __syncthreads();
    float M[8];
#pragma unroll
    for (int r = 0; r < 8; r++) {
        float mm = -3.0e38f;
#pragma unroll
        for (int ww = 0; ww < 8; ww++) mm = fmaxf(mm, redmax[ww * 16 + r + hi * 8]);
        M[r] = mm;
    }

    // exponentiate in place, accumulate row sums
    float ps[8];
#pragma unroll
    for (int r = 0; r < 8; r++) ps[r] = 0.0f;
#pragma unroll
    for (int nt = 0; nt < 16; nt++)
#pragma unroll
        for (int r = 0; r < 8; r++) {
            float e = __expf(accE[nt][r] - M[r]);
            accE[nt][r] = e;
            ps[r] += e;
        }
#pragma unroll
    for (int m = 1; m < 16; m <<= 1)
#pragma unroll
        for (int r = 0; r < 8; r++) ps[r] += __shfl_xor(ps[r], m, 32);
    if (lane16 == 0) {
#pragma unroll
        for (int r = 0; r < 8; r++) redsum[w * 16 + r + hi * 8] = ps[r];
    }
    __syncthreads();
    float inv[8];
#pragma unroll
    for (int r = 0; r < 8; r++) {
        float s = 0.0f;
#pragma unroll
        for (int ww = 0; ww < 8; ww++) s += redsum[ww * 16 + r + hi * 8];
        inv[r] = 1.0f / s;
    }

    // write normalized probabilities P[16][2048] bf16 to LDS
#pragma unroll
    for (int nt = 0; nt < 16; nt++) {
        int j = j0 + nt * 16 + lane16;
#pragma unroll
        for (int r = 0; r < 8; r++)
            P[(size_t)(r + hi * 8) * NN + j] = f2bf(accE[nt][r] * inv[r]);
    }
    __syncthreads();

    // ---------------- pass 2: out[i, c] = sum_j P[i,j] * v[c,j] ------------
    int c0 = w * 64;
    f32x8 accO[4];
#pragma unroll
    for (int nt = 0; nt < 4; nt++) accO[nt] = (f32x8){0,0,0,0,0,0,0,0};
    const uint16_t* vb = vA + (size_t)b * CC * NN;

    for (int jb = 0; jb < NN; jb += 32) {
        bf16x16 ap = load_a(P, NN, lane16, jb, hi);          // P rows from LDS
#pragma unroll
        for (int nt = 0; nt < 4; nt++) {
            bf16x16 bv_ = load_b(vb, NN, c0 + nt * 16 + lane16, jb, hi);
            accO[nt] = wmma_bf16(ap, bv_, accO[nt]);
        }
    }

    float g = gamma[0];
    float s = 1.0f / (1.0f + g);
    const float* xb = x   + (size_t)b * CC * NN;
    float*       ob = out + (size_t)b * CC * NN;
#pragma unroll
    for (int nt = 0; nt < 4; nt++) {
        int c = c0 + nt * 16 + lane16;     // D-frag col -> channel
#pragma unroll
        for (int r = 0; r < 8; r++) {
            int i = i0 + r + hi * 8;       // D-frag row -> position
            float xv = xb[(size_t)c * NN + i];
            ob[(size_t)c * NN + i] = (g * accO[nt][r] + xv) * s;
        }
    }
}

// ---------------------------------------------------------------------------
extern "C" void kernel_launch(void* const* d_in, const int* in_sizes, int n_in,
                              void* d_out, int out_size, void* d_ws, size_t ws_size,
                              hipStream_t stream) {
    const float* x     = (const float*)d_in[0];
    const float* Wq    = (const float*)d_in[1];
    const float* bq    = (const float*)d_in[2];
    const float* Wk    = (const float*)d_in[3];
    const float* bk    = (const float*)d_in[4];
    const float* Wv    = (const float*)d_in[5];
    const float* bv    = (const float*)d_in[6];
    const float* gamma = (const float*)d_in[7];
    float* out = (float*)d_out;

    // Workspace layout (bytes); total ~36.6 MB
    char* ws = (char*)d_ws;
    uint16_t* xT  = (uint16_t*)(ws);                         // B*N*C bf16 = 16 MB
    uint16_t* Wqb = (uint16_t*)(ws + 16777216);              // 64*512  = 64 KB
    uint16_t* Wkb = (uint16_t*)(ws + 16777216 + 65536);      // 64*512
    uint16_t* Wvb = (uint16_t*)(ws + 16777216 + 131072);     // 512*512 = 512 KB
    uint16_t* qA  = (uint16_t*)(ws + 17432576);              // B*N*64  = 2 MB
    uint16_t* kA  = (uint16_t*)(ws + 19529728);              // B*N*64  = 2 MB
    uint16_t* vA  = (uint16_t*)(ws + 21626880);              // B*C*N   = 16 MB

    // 1) weight conversion
    cvt_f32_bf16<<<(DD * CC + 255) / 256, 256, 0, stream>>>(Wq, Wqb, DD * CC);
    cvt_f32_bf16<<<(DD * CC + 255) / 256, 256, 0, stream>>>(Wk, Wkb, DD * CC);
    cvt_f32_bf16<<<(CC * CC + 255) / 256, 256, 0, stream>>>(Wv, Wvb, CC * CC);

    // 2) x -> xT bf16 (tiled transpose)
    transpose_x<<<dim3(NN / 32, CC / 32, BB), dim3(32, 8), 0, stream>>>(x, xT);

    // 3) projections (all WMMA bf16)
    proj_qk<<<BB * (NN / 64), 256, 0, stream>>>(xT, Wqb, bq, Wkb, bk, qA, kA);
    proj_v<<<BB * (CC / 64) * (NN / 128), 256, 0, stream>>>(xT, Wvb, bv, vA);

    // 4) fused attention (energy WMMA + block softmax + out WMMA)
    //    dynamic LDS: P 16*2048 bf16 (64KB) + 2*128 f32 reduction = 66560 B
    size_t lds = (size_t)16 * NN * 2 + 2 * 128 * sizeof(float);
    attn<<<BB * (NN / 16), 256, lds, stream>>>(qA, kA, vA, x, gamma, out);
}
